// AttentionBlock_12627203850356
// MI455X (gfx1250) — matrix operand order repaired and checked by reading, not compiled, not run on hardware
//
#include <hip/hip_runtime.h>
#include <hip/hip_bf16.h>

#define CDIM 512
#define LDIM 4096
#define BDIM 8
#define CPG 16

typedef __bf16 bf16t;
typedef __attribute__((ext_vector_type(4)))  float   v4f;
typedef __attribute__((ext_vector_type(4)))  int     v4i;
typedef __attribute__((ext_vector_type(4)))  __bf16  v4bf;
typedef __attribute__((ext_vector_type(8)))  __bf16  v8bf;
typedef __attribute__((ext_vector_type(16))) __bf16  v16bf;
typedef __attribute__((ext_vector_type(8)))  float   v8f;

union V16U { v16bf v; v8bf h[2]; };
union V8U  { v8bf v; bf16t e[8]; };
union V4U  { v4bf v; bf16t e[4]; };
union PK2  { unsigned int u; bf16t e[2]; };

#if __has_builtin(__builtin_amdgcn_global_load_async_to_lds_b128) && \
    __has_builtin(__builtin_amdgcn_s_wait_asynccnt)
#define HAVE_ASYNC_LDS 1
#define GAS __attribute__((address_space(1)))
#define LAS __attribute__((address_space(3)))
#else
#define HAVE_ASYNC_LDS 0
#endif

static __device__ __forceinline__ v8f wmma_bf16(v16bf a, v16bf b, v8f c) {
  // D(16x16,f32) = A(16x32,bf16) * B(32x16,bf16) + C
  return __builtin_amdgcn_wmma_f32_16x16x32_bf16(false, a, false, b, (short)0, c,
                                                 false, false);
}

// ---------------------------------------------------------------------------
// Kernel 0: one-time fp32 -> bf16 conversion of the four 512x512 weights.
// 65536 threads, each converts one float4 per matrix.
// ---------------------------------------------------------------------------
__global__ __launch_bounds__(256) void wcvt_kernel(
    const float* __restrict__ a, const float* __restrict__ b,
    const float* __restrict__ c, const float* __restrict__ d,
    bf16t* __restrict__ ao, bf16t* __restrict__ bo,
    bf16t* __restrict__ co, bf16t* __restrict__ dd) {
  const size_t i = (size_t)blockIdx.x * 256 + threadIdx.x;
  const float* src[4] = {a, b, c, d};
  bf16t* dst[4] = {ao, bo, co, dd};
#pragma unroll
  for (int m = 0; m < 4; ++m) {
    v4f x = *(const v4f*)(src[m] + i * 4);
    V4U p;
#pragma unroll
    for (int r = 0; r < 4; ++r) p.e[r] = (bf16t)x[r];
    *(v4bf*)(dst[m] + i * 4) = p.v;
  }
}

// ---------------------------------------------------------------------------
// Kernel 1: GroupNorm (fp32, (B,C,L)) -> bf16 h TRANSPOSED to (B, L, C).
// One block per (batch, group); group data is a contiguous 16*4096 fp32 run.
// Transpose goes through a padded LDS tile so reads stay coalesced and
// stores are 32-bit packed.
// ---------------------------------------------------------------------------
__global__ __launch_bounds__(256) void gn_kernel(const float* __restrict__ X,
                                                 const float* __restrict__ gamma,
                                                 const float* __restrict__ beta,
                                                 bf16t* __restrict__ Ht) {
  const int tid = threadIdx.x;
  const int lane = tid & 31, wave = tid >> 5;
  const int b = blockIdx.x >> 5;
  const int g = blockIdx.x & 31;
  const int c0 = g * CPG;
  const size_t base = ((size_t)b * CDIM + c0) * LDIM;  // contiguous 65536 floats

  __shared__ float redA[8], redB[8], stats[2];
  __shared__ float sg[CPG], sb[CPG];
  __shared__ __align__(16) bf16t tile[CPG][260];  // padded rows vs bank conflicts

  if (tid < CPG) { sg[tid] = gamma[c0 + tid]; sb[tid] = beta[c0 + tid]; }

  float s = 0.f, ss = 0.f;
  for (int i = tid; i < CPG * LDIM / 4; i += 256) {
    v4f x = *(const v4f*)(X + base + (size_t)i * 4);
#pragma unroll
    for (int r = 0; r < 4; ++r) { s += x[r]; ss += x[r] * x[r]; }
  }
  for (int m = 1; m < 32; m <<= 1) {
    s  += __shfl_xor(s,  m, 32);
    ss += __shfl_xor(ss, m, 32);
  }
  if (lane == 0) { redA[wave] = s; redB[wave] = ss; }
  __syncthreads();
  if (tid == 0) {
    float S = 0.f, SS = 0.f;
    for (int w = 0; w < 8; ++w) { S += redA[w]; SS += redB[w]; }
    const float invN = 1.f / (float)(CPG * LDIM);
    float mean = S * invN;
    float var = SS * invN - mean * mean;
    stats[0] = mean;
    stats[1] = rsqrtf(var + 1e-5f);
  }
  __syncthreads();
  const float mean = stats[0], rstd = stats[1];

  bf16t* hb = Ht + (size_t)b * LDIM * CDIM + c0;   // row l -> hb + l*CDIM + c
  for (int t = 0; t < LDIM / 256; ++t) {
    const int l0 = t * 256;
    __syncthreads();
    for (int i = tid; i < 1024; i += 256) {        // 16 ch x 256 l, 4 elems/iter
      const int c = i >> 6, l4 = (i & 63) * 4;
      v4f x = *(const v4f*)(X + base + (size_t)c * LDIM + l0 + l4);
      V4U p;
#pragma unroll
      for (int r = 0; r < 4; ++r)
        p.e[r] = (bf16t)((x[r] - mean) * rstd * sg[c] + sb[c]);
      *(v4bf*)&tile[c][l4] = p.v;
    }
    __syncthreads();
    for (int i = tid; i < 2048; i += 256) {        // packed 2-channel stores
      const int ll = i >> 3, cp = i & 7;
      PK2 p;
      p.e[0] = tile[2 * cp][ll];
      p.e[1] = tile[2 * cp + 1][ll];
      *reinterpret_cast<unsigned int*>(hb + (size_t)(l0 + ll) * CDIM + 2 * cp) = p.u;
    }
  }
}

// ---------------------------------------------------------------------------
// Kernel 2: q/k/v projections. h is (B,L,C) bf16; weights bf16.
// LDS-free: A and B operands are direct 16B global loads (L2-resident).
// grid (L/256, C/64, B*3); wave tile = 32x64 (2 M-tiles x 4 N-tiles).
// q,k written spatial-major (B,L,C); v written channel-major (B,C,L).
// q pre-scaled by 1/sqrt(C).
// ---------------------------------------------------------------------------
__global__ __launch_bounds__(256) void proj_kernel(
    const bf16t* __restrict__ ht,
    const bf16t* __restrict__ wqb, const bf16t* __restrict__ wkb,
    const bf16t* __restrict__ wvb,
    const float* __restrict__ bq, const float* __restrict__ bk,
    const float* __restrict__ bv,
    bf16t* __restrict__ qt, bf16t* __restrict__ kt, bf16t* __restrict__ v) {
  const int tid = threadIdx.x;
  const int lane = tid & 31, wave = tid >> 5;
  const int half = lane >> 4, l16 = lane & 15;
  const int b = blockIdx.z / 3;
  const int which = blockIdx.z % 3;
  const int m0 = blockIdx.y * 64 + (wave & 1) * 32;
  const int l0 = blockIdx.x * 256 + (wave >> 1) * 64;

  const bf16t* W    = (which == 0) ? wqb : (which == 1) ? wkb : wvb;
  const float* bias = (which == 0) ? bq  : (which == 1) ? bk  : bv;
  const bf16t* hb = ht + (size_t)b * LDIM * CDIM;

  v8f acc[2][4];
#pragma unroll
  for (int mt = 0; mt < 2; ++mt)
#pragma unroll
    for (int nt = 0; nt < 4; ++nt)
#pragma unroll
      for (int e = 0; e < 8; ++e) acc[mt][nt][e] = 0.f;

#pragma unroll 4
  for (int c0 = 0; c0 < CDIM; c0 += 32) {
    V16U av[2];
#pragma unroll
    for (int mt = 0; mt < 2; ++mt) {
      const bf16t* ap = W + (size_t)(m0 + mt * 16 + l16) * CDIM + c0;
      av[mt].h[0] = *(const v8bf*)(ap + half * 8);
      av[mt].h[1] = *(const v8bf*)(ap + 16 + half * 8);
    }
#pragma unroll
    for (int nt = 0; nt < 4; ++nt) {
      const bf16t* bp = hb + (size_t)(l0 + nt * 16 + l16) * CDIM + c0 + half * 8;
      V16U bvv;
      bvv.h[0] = *(const v8bf*)bp;
      bvv.h[1] = *(const v8bf*)(bp + 16);
#pragma unroll
      for (int mt = 0; mt < 2; ++mt)
        acc[mt][nt] = wmma_bf16(av[mt].v, bvv.v, acc[mt][nt]);
    }
  }

  if (which < 2) {
    // spatial-major (L, C): pack 8 consecutive channels into one 16B store
    bf16t* dst = ((which == 0) ? qt : kt) + (size_t)b * LDIM * CDIM;
    const float scale = (which == 0) ? 0.04419417382415922f : 1.0f;  // 1/sqrt(512)
#pragma unroll
    for (int mt = 0; mt < 2; ++mt)
#pragma unroll
      for (int nt = 0; nt < 4; ++nt) {
        V8U p;
#pragma unroll
        for (int r = 0; r < 8; ++r) {
          const int m = m0 + mt * 16 + half * 8 + r;
          p.e[r] = (bf16t)((acc[mt][nt][r] + bias[m]) * scale);
        }
        *(v8bf*)(dst + (size_t)(l0 + nt * 16 + l16) * CDIM + m0 + mt * 16 + half * 8) = p.v;
      }
  } else {
    // v: channel-major (C, L)
    bf16t* dst = v + (size_t)b * CDIM * LDIM;
#pragma unroll
    for (int mt = 0; mt < 2; ++mt)
#pragma unroll
      for (int nt = 0; nt < 4; ++nt)
#pragma unroll
        for (int r = 0; r < 8; ++r) {
          const int m = m0 + mt * 16 + half * 8 + r;
          dst[(size_t)m * LDIM + l0 + nt * 16 + l16] = (bf16t)(acc[mt][nt][r] + bias[m]);
        }
  }
}

// ---------------------------------------------------------------------------
// Kernel 3: fused flash attention.
// One workgroup (8 waves) = one batch x 32-query block.
// Q block (contiguous 32KB of q_t) resident in LDS (async-to-LDS staged).
// S phase: barrier-free; B operands straight from global k_t (L,C).
// Online softmax across waves via LDS; P transposed through LDS for P*V.
// Output attn_t is (B, L, C) bf16 with 16B packed stores.
// ---------------------------------------------------------------------------
__global__ __launch_bounds__(256) void attn_kernel(const bf16t* __restrict__ qt,
                                                   const bf16t* __restrict__ kt,
                                                   const bf16t* __restrict__ vmat,
                                                   bf16t* __restrict__ ot) {
  const int tid = threadIdx.x;
  const int lane = tid & 31, wave = tid >> 5;
  const int half = lane >> 4, l16 = lane & 15;
  const int q0 = blockIdx.x * 32;
  const int b = blockIdx.y;

  const bf16t* qb = qt + ((size_t)b * LDIM + q0) * CDIM;   // contiguous 32x512
  const bf16t* kb = kt + (size_t)b * LDIM * CDIM;
  const bf16t* vb = vmat + (size_t)b * CDIM * LDIM;
  bf16t* ob = ot + (size_t)b * LDIM * CDIM;

  __shared__ __align__(16) bf16t Qs[32][CDIM];   // 32 KB
  __shared__ __align__(16) bf16t Ps[32][128];    // 8 KB, P (query, key)
  __shared__ float redmax[32][8];
  __shared__ float redsum[32][8];
  __shared__ float run_max[32], run_sum[32], qsc[32];

  // ---- stage Q block (straight 32KB memcpy into LDS) ----
#if HAVE_ASYNC_LDS
  for (int i = tid; i < 2048; i += 256) {
    __builtin_amdgcn_global_load_async_to_lds_b128(
        (GAS v4i*)(qb + (size_t)i * 8),
        (LAS v4i*)(&Qs[0][0] + (size_t)i * 8),
        0, 0);
  }
  __builtin_amdgcn_s_wait_asynccnt(0);
#else
  for (int i = tid; i < 2048; i += 256)
    *((v8bf*)&Qs[0][0] + i) = *((const v8bf*)qb + i);
#endif
  if (tid < 32) { run_max[tid] = -1e30f; run_sum[tid] = 0.f; }

  const int cw0 = wave * 64;   // this wave's output-channel slice (PV phase)
  const int kw0 = wave * 16;   // this wave's key slice within the block

  v8f oacc[4][2];
#pragma unroll
  for (int ct = 0; ct < 4; ++ct)
#pragma unroll
    for (int nt = 0; nt < 2; ++nt)
#pragma unroll
      for (int e = 0; e < 8; ++e) oacc[ct][nt][e] = 0.f;

  __syncthreads();

  for (int kb0 = 0; kb0 < LDIM; kb0 += 128) {
    // ------------- Phase A: S = Q^T K, barrier-free ----------------------
    v8f sacc[2];
#pragma unroll
    for (int mt = 0; mt < 2; ++mt)
#pragma unroll
      for (int e = 0; e < 8; ++e) sacc[mt][e] = 0.f;

    const bf16t* krow = kb + (size_t)(kb0 + kw0 + l16) * CDIM;
#pragma unroll 4
    for (int c0 = 0; c0 < CDIM; c0 += 32) {
      V16U bv_;
      bv_.h[0] = *(const v8bf*)(krow + c0 + half * 8);
      bv_.h[1] = *(const v8bf*)(krow + c0 + 16 + half * 8);
#pragma unroll
      for (int mt = 0; mt < 2; ++mt) {
        V16U av;
        av.h[0] = *(const v8bf*)&Qs[mt * 16 + l16][c0 + half * 8];
        av.h[1] = *(const v8bf*)&Qs[mt * 16 + l16][c0 + 16 + half * 8];
        sacc[mt] = wmma_bf16(av.v, bv_.v, sacc[mt]);
      }
    }

    // ------------- Phase B: online softmax (cross-wave via LDS) ----------
#pragma unroll
    for (int mt = 0; mt < 2; ++mt)
#pragma unroll
      for (int r = 0; r < 8; ++r) {
        float x = sacc[mt][r];
        x = fmaxf(x, __shfl_xor(x, 1, 32));
        x = fmaxf(x, __shfl_xor(x, 2, 32));
        x = fmaxf(x, __shfl_xor(x, 4, 32));
        x = fmaxf(x, __shfl_xor(x, 8, 32));
        if (l16 == 0) redmax[mt * 16 + half * 8 + r][wave] = x;
      }
    __syncthreads();
    if (tid < 32) {
      float m = redmax[tid][0];
      for (int w = 1; w < 8; ++w) m = fmaxf(m, redmax[tid][w]);
      float nm = fmaxf(run_max[tid], m);
      float sc = __expf(run_max[tid] - nm);
      run_max[tid] = nm;
      qsc[tid] = sc;
      run_sum[tid] *= sc;
    }
    __syncthreads();
#pragma unroll
    for (int mt = 0; mt < 2; ++mt)
#pragma unroll
      for (int r = 0; r < 8; ++r) {
        const int row = mt * 16 + half * 8 + r;
        float p = __expf(sacc[mt][r] - run_max[row]);
        Ps[row][kw0 + l16] = (bf16t)p;
        float su = p;
        su += __shfl_xor(su, 1, 32);
        su += __shfl_xor(su, 2, 32);
        su += __shfl_xor(su, 4, 32);
        su += __shfl_xor(su, 8, 32);
        if (l16 == 0) redsum[row][wave] = su;
      }
    __syncthreads();
    if (tid < 32) {
      float su = 0.f;
      for (int w = 0; w < 8; ++w) su += redsum[tid][w];
      run_sum[tid] += su;
    }

    // ------------- Phase C: O = O*scale + V P^T --------------------------
    const float sc0 = qsc[l16];
    const float sc1 = qsc[16 + l16];
#pragma unroll
    for (int ct = 0; ct < 4; ++ct)
#pragma unroll
      for (int e = 0; e < 8; ++e) {
        oacc[ct][0][e] *= sc0;
        oacc[ct][1][e] *= sc1;
      }

#pragma unroll
    for (int k0 = 0; k0 < 128; k0 += 32) {
#pragma unroll
      for (int ct = 0; ct < 4; ++ct) {
        const int c = cw0 + ct * 16 + l16;
        const bf16t* vp = vb + (size_t)c * LDIM + kb0 + k0;
        V16U av;
        av.h[0] = *(const v8bf*)(vp + half * 8);
        av.h[1] = *(const v8bf*)(vp + 16 + half * 8);
#pragma unroll
        for (int nt = 0; nt < 2; ++nt) {
          V16U bv_;
          bv_.h[0] = *(const v8bf*)&Ps[nt * 16 + l16][k0 + half * 8];
          bv_.h[1] = *(const v8bf*)&Ps[nt * 16 + l16][k0 + 16 + half * 8];
          oacc[ct][nt] = wmma_bf16(av.v, bv_.v, oacc[ct][nt]);
        }
      }
    }
  }

  __syncthreads();
#pragma unroll
  for (int nt = 0; nt < 2; ++nt) {
    const float inv = 1.f / run_sum[nt * 16 + l16];
#pragma unroll
    for (int ct = 0; ct < 4; ++ct) {
      V8U p;
#pragma unroll
      for (int r = 0; r < 8; ++r) p.e[r] = (bf16t)(oacc[ct][nt][r] * inv);
      *(v8bf*)(ob + (size_t)(q0 + nt * 16 + l16) * CDIM + cw0 + ct * 16 + half * 8) = p.v;
    }
  }
}

// ---------------------------------------------------------------------------
// Kernel 4: out = x + wo * attn + bo   (fp32 (B,C,L) output, LDS-free)
// ---------------------------------------------------------------------------
__global__ __launch_bounds__(256) void oproj_kernel(const bf16t* __restrict__ at,
                                                    const bf16t* __restrict__ wob,
                                                    const float* __restrict__ bo,
                                                    const float* __restrict__ x,
                                                    float* __restrict__ out) {
  const int tid = threadIdx.x;
  const int lane = tid & 31, wave = tid >> 5;
  const int half = lane >> 4, l16 = lane & 15;
  const int b = blockIdx.z;
  const int m0 = blockIdx.y * 64 + (wave & 1) * 32;
  const int l0 = blockIdx.x * 256 + (wave >> 1) * 64;

  const bf16t* ab = at + (size_t)b * LDIM * CDIM;

  v8f acc[2][4];
#pragma unroll
  for (int mt = 0; mt < 2; ++mt)
#pragma unroll
    for (int nt = 0; nt < 4; ++nt)
#pragma unroll
      for (int e = 0; e < 8; ++e) acc[mt][nt][e] = 0.f;

#pragma unroll 4
  for (int c0 = 0; c0 < CDIM; c0 += 32) {
    V16U av[2];
#pragma unroll
    for (int mt = 0; mt < 2; ++mt) {
      const bf16t* ap = wob + (size_t)(m0 + mt * 16 + l16) * CDIM + c0;
      av[mt].h[0] = *(const v8bf*)(ap + half * 8);
      av[mt].h[1] = *(const v8bf*)(ap + 16 + half * 8);
    }
#pragma unroll
    for (int nt = 0; nt < 4; ++nt) {
      const bf16t* bp = ab + (size_t)(l0 + nt * 16 + l16) * CDIM + c0 + half * 8;
      V16U bvv;
      bvv.h[0] = *(const v8bf*)bp;
      bvv.h[1] = *(const v8bf*)(bp + 16);
#pragma unroll
      for (int mt = 0; mt < 2; ++mt)
        acc[mt][nt] = wmma_bf16(av[mt].v, bvv.v, acc[mt][nt]);
    }
  }

#pragma unroll
  for (int mt = 0; mt < 2; ++mt)
#pragma unroll
    for (int nt = 0; nt < 4; ++nt)
#pragma unroll
      for (int r = 0; r < 8; ++r) {
        const int m = m0 + mt * 16 + half * 8 + r;
        const size_t idx = ((size_t)b * CDIM + m) * LDIM + l0 + nt * 16 + l16;
        out[idx] = x[idx] + acc[mt][nt][r] + bo[m];
      }
}

// ---------------------------------------------------------------------------
extern "C" void kernel_launch(void* const* d_in, const int* in_sizes, int n_in,
                              void* d_out, int out_size, void* d_ws, size_t ws_size,
                              hipStream_t stream) {
  (void)in_sizes; (void)n_in; (void)out_size; (void)ws_size;
  const float* x   = (const float*)d_in[0];
  const float* gam = (const float*)d_in[1];
  const float* bet = (const float*)d_in[2];
  const float* wq  = (const float*)d_in[3];
  const float* bq  = (const float*)d_in[4];
  const float* wk  = (const float*)d_in[5];
  const float* bk  = (const float*)d_in[6];
  const float* wv  = (const float*)d_in[7];
  const float* bv  = (const float*)d_in[8];
  const float* wo  = (const float*)d_in[9];
  const float* bo  = (const float*)d_in[10];
  float* out = (float*)d_out;

  // Workspace (bf16): h_t | q_t | k_t | v | attn_t | 4 weight copies (~162 MB)
  const size_t n = (size_t)BDIM * CDIM * LDIM;
  const size_t wn = (size_t)CDIM * CDIM;
  bf16t* ht  = (bf16t*)d_ws;
  bf16t* qt  = ht + n;
  bf16t* kt  = qt + n;
  bf16t* vv  = kt + n;
  bf16t* at  = vv + n;
  bf16t* wqb = at + n;
  bf16t* wkb = wqb + wn;
  bf16t* wvb = wkb + wn;
  bf16t* wob = wvb + wn;

  wcvt_kernel<<<dim3(wn / 4 / 256), dim3(256), 0, stream>>>(
      wq, wk, wv, wo, wqb, wkb, wvb, wob);
  gn_kernel<<<dim3(BDIM * 32), dim3(256), 0, stream>>>(x, gam, bet, ht);
  proj_kernel<<<dim3(LDIM / 256, CDIM / 64, BDIM * 3), dim3(256), 0, stream>>>(
      ht, wqb, wkb, wvb, bq, bk, bv, qt, kt, vv);
  attn_kernel<<<dim3(LDIM / 32, BDIM), dim3(256), 0, stream>>>(qt, kt, vv, at);
  oproj_kernel<<<dim3(LDIM / 256, CDIM / 64, BDIM), dim3(256), 0, stream>>>(
      at, wob, bo, x, out);
}
